// KANFFN_36472862277821
// MI455X (gfx1250) — matrix operand order repaired and checked by reading, not compiled, not run on hardware
//
#include <hip/hip_runtime.h>
#include <hip/hip_bf16.h>

// ---------------------------------------------------------------------------
// KAN FFN (1024 -> 128 -> 1024) on gfx1250, f16 WMMA with f32 accumulate.
//
// Per-scalar expansion to 8 features turns each KAN layer into a single GEMM:
//   feat[k=i*8+g] : g=0..5 cubic B-spline bases (uniform knots), g=6 silu(x),
//   g=7 zero pad.  Weights packed as Wt[o][i*8+g] in f16.
// ---------------------------------------------------------------------------

typedef __attribute__((ext_vector_type(16))) _Float16 v16h;
typedef __attribute__((ext_vector_type(8)))  _Float16 v8h;
typedef __attribute__((ext_vector_type(8)))  float    v8f;

#define D_MODEL    1024
#define KAN_HIDDEN 128
#define NTOK       16384

#define M_TILE 64        // tokens per block
#define N_TILE 128       // outputs per block
#define DI     16        // input dims per K-chunk
#define KC     (DI * 8)  // 128 f16 K elements per chunk
#define STR    (KC + 8)  // padded LDS row stride (halves): 272B rows, 16B aligned

// Uniform cubic B-spline expansion. Knots t_j = -1 + (j-3)*(2/3), j=0..9.
// For x in [t_j, t_{j+1}) only bases i=j-3..j are nonzero with the cardinal
// cubic blending weights; bases outside 0..5 are dropped (matches reference).
__device__ __forceinline__ v8h kan_expand8(float x) {
  float sil = x / (1.0f + __expf(-x));     // silu base path
  float t  = (x + 3.0f) * 1.5f;            // (x - t0)/h, h = 2/3
  float fj = floorf(t);
  int   j  = (int)fj;
  float u  = t - fj;
  float u2 = u * u, u3 = u2 * u;
  float um = 1.0f - u;
  float w0 = um * um * um * (1.0f / 6.0f);
  float w1 = (3.0f * u3 - 6.0f * u2 + 4.0f) * (1.0f / 6.0f);
  float w2 = (-3.0f * u3 + 3.0f * u2 + 3.0f * u + 1.0f) * (1.0f / 6.0f);
  float w3 = u3 * (1.0f / 6.0f);
  bool inr = (t >= 0.0f) && (t < 9.0f);    // x in [-3, 3): spline support
  if (!inr) { w0 = 0.0f; w1 = 0.0f; w2 = 0.0f; w3 = 0.0f; }
  v8h r;
#pragma unroll
  for (int i = 0; i < 6; ++i) {
    int d = i + 3 - j;                     // which of w0..w3 lands on basis i
    float v = (d == 0) ? w0 : (d == 1) ? w1 : (d == 2) ? w2 : (d == 3) ? w3 : 0.0f;
    r[i] = (_Float16)v;
  }
  r[6] = (_Float16)sil;
  r[7] = (_Float16)0.0f;
  return r;
}

// Build a 16-half WMMA fragment from two aligned 8-half LDS reads.
// A-matrix frag: hi_off = 16 (K and K+16 halves).  B-matrix frag: hi_off = 16
// (same K and K+16 halves as A).
__device__ __forceinline__ v16h ld_frag(const _Float16* p, int hi_off) {
  v8h lo = *(const v8h*)p;
  v8h hi = *(const v8h*)(p + hi_off);
  v16h r;
#pragma unroll
  for (int i = 0; i < 8; ++i) { r[i] = lo[i]; r[i + 8] = hi[i]; }
  return r;
}

// Pack weights: Wt[o][i*8+g] = g<6 ? coef[i][o][g]*scale_sp[i][o]
//                            : g==6 ? scale_base[i][o] : 0
__global__ __launch_bounds__(256) void kan_pack_weights(
    const float* __restrict__ coef,        // [in][out][6]
    const float* __restrict__ scale_base,  // [in][out]
    const float* __restrict__ scale_sp,    // [in][out]
    _Float16* __restrict__ wt,             // [out][in*8]
    int in_dim, int out_dim) {
  int t = blockIdx.x * 256 + threadIdx.x;
  if (t >= in_dim * out_dim) return;
  int i = t / out_dim;
  int o = t - i * out_dim;
  const float* c = coef + (size_t)t * 6;
  float ssp = scale_sp[t];
  float sb  = scale_base[t];
  v8h v;
#pragma unroll
  for (int g = 0; g < 6; ++g) v[g] = (_Float16)(c[g] * ssp);
  v[6] = (_Float16)sb;
  v[7] = (_Float16)0.0f;
  *(v8h*)&wt[(size_t)o * (size_t)(in_dim * 8) + (size_t)i * 8] = v;
}

// Fused expand + GEMM for one KAN layer.
//   act : [NTOK][IN]  f32
//   wt  : [n_out][IN*8] f16 (packed)
//   out : [NTOK][n_out] f32
template <int IN>
__global__ __launch_bounds__(256) void kan_layer_gemm(
    const float* __restrict__ act,
    const _Float16* __restrict__ wt,
    float* __restrict__ out,
    int n_out) {
  __shared__ _Float16 As[M_TILE * STR];   // 64 tokens  x 128(+8) halves
  __shared__ _Float16 Bs[N_TILE * STR];   // 128 outputs x 128(+8) halves

  const int tid  = threadIdx.x;
  const int lane = tid & 31;
  const int wid  = tid >> 5;     // 8 waves
  const int wm   = wid >> 2;     // 0..1 -> 32-row slab
  const int wn   = wid & 3;      // 0..3 -> 32-col slab
  const int m0   = blockIdx.x * M_TILE;
  const int n0   = blockIdx.y * N_TILE;
  const int lrow = lane & 15;
  const int lhi  = lane >> 4;

  v8f acc[2][2] = {};

  for (int c = 0; c < IN / DI; ++c) {
    const int d0 = c * DI;
    __syncthreads();
    // --- expand activation chunk into As: 64 tok x 16 dims -> 8 halves each
#pragma unroll
    for (int e = tid; e < M_TILE * DI; e += 256) {
      const int tok = e >> 4;          // / DI
      const int dm  = e & (DI - 1);
      const float xv = act[(size_t)(m0 + tok) * IN + (d0 + dm)];
      *(v8h*)&As[tok * STR + dm * 8] = kan_expand8(xv);
    }
    // --- load weight tile into Bs: 128 cols x 128 halves (L2-resident)
    {
      const int o  = tid >> 1;
      const int kh = (tid & 1) * 64;
      const uint4* src = (const uint4*)(wt + (size_t)(n0 + o) * (size_t)(IN * 8)
                                           + (size_t)d0 * 8 + kh);
      uint4* dst = (uint4*)&Bs[o * STR + kh];
#pragma unroll
      for (int q = 0; q < 4; ++q) dst[q] = src[q];
    }
    __syncthreads();
    // --- 4 WMMA K-steps (K=32 each) over this chunk
#pragma unroll
    for (int kk = 0; kk < KC / 32; ++kk) {
      const int ka = kk * 32 + lhi * 8;    // A: K / K+16 half-pairs per lane
      const int kb = kk * 32 + lhi * 8;    // B: K / K+16 half-pairs per lane
      v16h a0 = ld_frag(&As[(wm * 32 + lrow) * STR + ka], 16);
      v16h a1 = ld_frag(&As[(wm * 32 + 16 + lrow) * STR + ka], 16);
      v16h b0 = ld_frag(&Bs[(wn * 32 + lrow) * STR + kb], 16);
      v16h b1 = ld_frag(&Bs[(wn * 32 + 16 + lrow) * STR + kb], 16);
      acc[0][0] = __builtin_amdgcn_wmma_f32_16x16x32_f16(false, a0, false, b0,
                                                         (short)0, acc[0][0], false, false);
      acc[0][1] = __builtin_amdgcn_wmma_f32_16x16x32_f16(false, a0, false, b1,
                                                         (short)0, acc[0][1], false, false);
      acc[1][0] = __builtin_amdgcn_wmma_f32_16x16x32_f16(false, a1, false, b0,
                                                         (short)0, acc[1][0], false, false);
      acc[1][1] = __builtin_amdgcn_wmma_f32_16x16x32_f16(false, a1, false, b1,
                                                         (short)0, acc[1][1], false, false);
    }
  }

  // --- store C: v8f layout -> M = lhi*8 + r, N = lrow (per ISA C/D table)
#pragma unroll
  for (int mb = 0; mb < 2; ++mb)
#pragma unroll
    for (int nb = 0; nb < 2; ++nb) {
      const int row = m0 + wm * 32 + mb * 16 + lhi * 8;
      const int col = n0 + wn * 32 + nb * 16 + lrow;
      float* op = out + (size_t)row * n_out + col;
#pragma unroll
      for (int r = 0; r < 8; ++r) op[(size_t)r * n_out] = acc[mb][nb][r];
    }
}

extern "C" void kernel_launch(void* const* d_in, const int* in_sizes, int n_in,
                              void* d_out, int out_size, void* d_ws, size_t ws_size,
                              hipStream_t stream) {
  (void)in_sizes; (void)n_in; (void)out_size; (void)ws_size;
  const float* x           = (const float*)d_in[0];
  const float* coef1       = (const float*)d_in[1];
  const float* scale_base1 = (const float*)d_in[2];
  const float* scale_sp1   = (const float*)d_in[3];
  const float* coef2       = (const float*)d_in[4];
  const float* scale_base2 = (const float*)d_in[5];
  const float* scale_sp2   = (const float*)d_in[6];
  float* out = (float*)d_out;

  char* ws = (char*)d_ws;
  _Float16* Wt1 = (_Float16*)(ws);                 // 128  x 8192 f16 = 2 MiB
  _Float16* Wt2 = (_Float16*)(ws + (2u << 20));    // 1024 x 1024 f16 = 2 MiB
  float*    y1  = (float*)   (ws + (4u << 20));    // 16384 x 128 f32 = 8 MiB

  kan_pack_weights<<<(D_MODEL * KAN_HIDDEN + 255) / 256, 256, 0, stream>>>(
      coef1, scale_base1, scale_sp1, Wt1, D_MODEL, KAN_HIDDEN);
  kan_pack_weights<<<(KAN_HIDDEN * D_MODEL + 255) / 256, 256, 0, stream>>>(
      coef2, scale_base2, scale_sp2, Wt2, KAN_HIDDEN, D_MODEL);

  // Layer 1: 1024 -> 128.  N-tile covers all 128 outputs, so each x element
  // is expanded exactly once.
  kan_layer_gemm<D_MODEL><<<dim3(NTOK / M_TILE, KAN_HIDDEN / N_TILE), 256, 0, stream>>>(
      x, Wt1, y1, KAN_HIDDEN);
  // Layer 2: 128 -> 1024.
  kan_layer_gemm<KAN_HIDDEN><<<dim3(NTOK / M_TILE, D_MODEL / N_TILE), 256, 0, stream>>>(
      y1, Wt2, out, D_MODEL);
}
